// VanillaRNN_73512660239039
// MI455X (gfx1250) — matrix operand order repaired and checked by reading, not compiled, not run on hardware
//
#include <hip/hip_runtime.h>

typedef __attribute__((ext_vector_type(16))) _Float16 v16h;
typedef __attribute__((ext_vector_type(8)))  _Float16 v8h;
typedef __attribute__((ext_vector_type(8)))  float    v8f;
typedef __attribute__((ext_vector_type(4)))  float    v4f;

#define T_SEQ  2048
#define HID    128
#define NBATCH 8192
#define NCLS   10
#define NT     32      // batch columns per workgroup (2 WMMA N-tiles)
#define NTHR   256     // 8 waves

// gfx1250 has V_TANH_F32 (TRANS op, co-executes with WMMA). Fall back to the
// exp-based form if this toolchain lacks the builtin.
#if __has_builtin(__builtin_amdgcn_tanhf)
__device__ __forceinline__ float fast_tanh(float v) {
    return __builtin_amdgcn_tanhf(v);
}
#else
__device__ __forceinline__ float fast_tanh(float v) {
    float e = __expf(2.0f * v);
    return 1.0f - 2.0f * __builtin_amdgcn_rcpf(e + 1.0f);
}
#endif

union HPack {
    v8h          v;
    unsigned int u[4];
};

__global__ __launch_bounds__(NTHR)
void rnn_wmma_kernel(const float* __restrict__ x,
                     const float* __restrict__ w_hx,
                     const float* __restrict__ w_hh,
                     const float* __restrict__ b_h,
                     const float* __restrict__ w_ph,
                     const float* __restrict__ b_p,
                     float* __restrict__ out) {
    // double-buffered h^T in LDS: [n][k] f16, 2 * 32 * 128 * 2B = 16 KB
    __shared__ _Float16 hbuf[2][NT * HID];

    const int tid   = threadIdx.x;
    const int wave  = tid >> 5;
    const int lane  = tid & 31;
    const int l16   = lane & 15;
    const int lhalf = lane >> 4;       // 0: K/M-low half, 1: high half
    const int roff  = lhalf * 8;       // C/D: VGPR r holds row M = roff + r
    const int mbase = wave * 16;       // this wave's 16 rows of W_hh
    const int batch_base = blockIdx.x * NT;

    // ---- A operands: this wave's 16x128 slice of W_hh, f32->f16, resident in VGPRs.
    // 16-bit A 16x32 layout (matches B): lane(l16,half): elements 0..15 = K {kb+half*16+0..15}
    v16h a[4];
    {
        const float* wrow = w_hh + (size_t)(mbase + l16) * HID;
        #pragma unroll
        for (int kc = 0; kc < 4; ++kc) {
            #pragma unroll
            for (int j = 0; j < 16; ++j) {
                int k = kc * 32 + lhalf * 16 + j;
                a[kc][j] = (_Float16)wrow[k];
            }
        }
    }

    // per-lane row constants for the 8 output rows this lane accumulates
    float whx_r[8], bh_r[8];
    #pragma unroll
    for (int r = 0; r < 8; ++r) {
        whx_r[r] = w_hx[mbase + roff + r];
        bh_r[r]  = b_h [mbase + roff + r];
    }

    // h0 = 0
    for (int i = tid; i < NT * HID; i += NTHR) hbuf[0][i] = (_Float16)0.0f;
    __syncthreads();

    // x streams: this lane's two batch columns (one per N-tile)
    const size_t ga = (size_t)(batch_base + l16)      * T_SEQ;
    const size_t gb = (size_t)(batch_base + 16 + l16) * T_SEQ;

    int p = 0;
    for (int t4 = 0; t4 < T_SEQ / 4; ++t4) {
        v4f xa = *(const v4f*)(x + ga + t4 * 4);
        v4f xb = *(const v4f*)(x + gb + t4 * 4);
        #pragma unroll
        for (int j = 0; j < 4; ++j) {
            const _Float16* hc = hbuf[p];
            float xs_a = xa[j], xs_b = xb[j];

            // fold bias + input contribution into the C operand:
            // acc[r] = b_h[m] + w_hx[m] * x  (per row m = mbase+roff+r)
            v8f acc0, acc1;
            #pragma unroll
            for (int r = 0; r < 8; ++r) {
                acc0[r] = fmaf(whx_r[r], xs_a, bh_r[r]);
                acc1[r] = fmaf(whx_r[r], xs_b, bh_r[r]);
            }

            // B (32x16 KxN) chunk: lanes 0-15 K=kb..kb+15 of col N=l16,
            // lanes 16-31 K=kb+16..kb+31 -> contiguous 32B in h^T[n][k]
            #pragma unroll
            for (int kc = 0; kc < 4; ++kc) {
                v16h b0 = *(const v16h*)(hc + l16 * HID        + kc * 32 + lhalf * 16);
                v16h b1 = *(const v16h*)(hc + (16 + l16) * HID + kc * 32 + lhalf * 16);
                acc0 = __builtin_amdgcn_wmma_f32_16x16x32_f16(false, a[kc], false, b0,
                                                              (short)0, acc0, false, false);
                acc1 = __builtin_amdgcn_wmma_f32_16x16x32_f16(false, a[kc], false, b1,
                                                              (short)0, acc1, false, false);
            }

            // epilogue: h = tanh(acc) (v_tanh_f32, TRANS), pack pairs with
            // v_cvt_pkrtz_f16_f32 into the 16B ds_store payload
            HPack h0, h1;
            #pragma unroll
            for (int r = 0; r < 8; r += 2) {
                float t0 = fast_tanh(acc0[r]);
                float t1 = fast_tanh(acc0[r + 1]);
                float s0 = fast_tanh(acc1[r]);
                float s1 = fast_tanh(acc1[r + 1]);
                h0.u[r >> 1] = __builtin_bit_cast(unsigned int,
                                   __builtin_amdgcn_cvt_pkrtz(t0, t1));
                h1.u[r >> 1] = __builtin_bit_cast(unsigned int,
                                   __builtin_amdgcn_cvt_pkrtz(s0, s1));
            }

            // D rows roff..roff+7 are contiguous k in h^T -> one 16B store per tile
            _Float16* hn = hbuf[p ^ 1];
            *(v8h*)(hn + l16 * HID        + mbase + roff) = h0.v;
            *(v8h*)(hn + (16 + l16) * HID + mbase + roff) = h1.v;
            __syncthreads();
            p ^= 1;
        }
    }

    // final projection: out[b, c] = w_ph[c,:] . h_last[:, b] + b_p[c]
    const _Float16* hl = hbuf[p];
    for (int idx = tid; idx < NCLS * NT; idx += NTHR) {
        int c = idx >> 5;          // 0..9
        int n = idx & 31;          // 0..31
        const _Float16* hv = hl + n * HID;
        const float*    wr = w_ph + c * HID;
        float s = b_p[c];
        #pragma unroll 8
        for (int k = 0; k < HID; ++k) s = fmaf(wr[k], (float)hv[k], s);
        out[(size_t)(batch_base + n) * NCLS + c] = s;
    }
}

extern "C" void kernel_launch(void* const* d_in, const int* in_sizes, int n_in,
                              void* d_out, int out_size, void* d_ws, size_t ws_size,
                              hipStream_t stream) {
    const float* x    = (const float*)d_in[0];
    const float* w_hx = (const float*)d_in[1];
    const float* w_hh = (const float*)d_in[2];
    const float* b_h  = (const float*)d_in[3];
    const float* w_ph = (const float*)d_in[4];
    const float* b_p  = (const float*)d_in[5];
    float* out = (float*)d_out;

    dim3 grid(NBATCH / NT);   // 256 workgroups
    dim3 block(NTHR);         // 8 waves of 32
    rnn_wmma_kernel<<<grid, block, 0, stream>>>(x, w_hx, w_hh, b_h, w_ph, b_p, out);
}
